// LDGCNN_17849884082380
// MI455X (gfx1250) — matrix operand order repaired and checked by reading, not compiled, not run on hardware
//
#include <hip/hip_runtime.h>

// Problem constants (match reference)
#define BB     16
#define NN     2048
#define KK     20
#define CAT_C  512      // 64 + 64 + 128 + 256

typedef __attribute__((ext_vector_type(16))) _Float16 v16h;
typedef __attribute__((ext_vector_type(8)))  float    v8f;

__device__ __forceinline__ float lrelu(float v) { return v >= 0.f ? v : 0.2f * v; }

// ---------------------------------------------------------------------------
// Pack W (co x ci, f32) -> f16 B-fragments for v_wmma_f32_16x16x32_f16.
// Fragment layout (ISA 7.12.2, 16-bit B 32x16): lane L holds column
// n = ct*16 + (L&15), halves j=0..7 hold K = kt*32 + (L>>4)*8 + j, j=8..15 hold K = kt*32 + 16 + (L>>4)*8 + (j-8).
// Stored contiguously: ((kt*numCT + ct)*32 + lane)*16 + j.
// BN gamma folded into the weight rows when g != nullptr.
// ---------------------------------------------------------------------------
__global__ void pack_weight(const float* __restrict__ W, const float* __restrict__ g,
                            _Float16* __restrict__ dst, int ci, int co)
{
    int t = blockIdx.x * blockDim.x + threadIdx.x;
    if (t >= ci * co) return;
    int j    = t & 15;
    int lane = (t >> 4) & 31;
    int tile = t >> 9;              // kt*numCT + ct
    int numCT = co >> 4;
    int kt = tile / numCT;
    int ct = tile % numCT;
    int n = (ct << 4) + (lane & 15);
    int k = (kt << 5) + ((lane >> 4) << 3) + ((j >> 3) << 4) + (j & 7);
    float w = W[(size_t)n * ci + k];
    if (g) w *= g[n];
    dst[t] = (_Float16)w;
}

// ---------------------------------------------------------------------------
// EdgeConv: h0[p,o] = max_k lrelu(g*(W_edge . [nb-ctr, ctr]) + be), 6->64.
// One thread per (point, 8-output group). Writes f16 into cat[:, 0:64].
// ---------------------------------------------------------------------------
__global__ void edge_conv(const float* __restrict__ x, const long long* __restrict__ idx,
                          const float* __restrict__ w, const float* __restrict__ g,
                          const float* __restrict__ be, _Float16* __restrict__ cat)
{
    int t = blockIdx.x * blockDim.x + threadIdx.x;
    if (t >= BB * NN * 8) return;
    int p  = t >> 3;                 // b*N + n
    int og = (t & 7) << 3;           // output-channel group base
    int b  = p / NN;
    const float cx = x[(size_t)p * 3 + 0];
    const float cy = x[(size_t)p * 3 + 1];
    const float cz = x[(size_t)p * 3 + 2];
    float wr[8][6], gg[8], bbv[8], acc[8];
#pragma unroll
    for (int o = 0; o < 8; ++o) {
#pragma unroll
        for (int c = 0; c < 6; ++c) wr[o][c] = w[(og + o) * 6 + c];
        gg[o] = g[og + o]; bbv[o] = be[og + o];
        acc[o] = -1e30f;
    }
    const long long* ip = idx + (size_t)p * KK;
    for (int k = 0; k < KK; ++k) {
        int j = (int)ip[k];
        const float* xn = x + ((size_t)b * NN + j) * 3;
        float d0 = xn[0] - cx, d1 = xn[1] - cy, d2 = xn[2] - cz;
#pragma unroll
        for (int o = 0; o < 8; ++o) {
            float s = wr[o][0] * d0 + wr[o][1] * d1 + wr[o][2] * d2
                    + wr[o][3] * cx + wr[o][4] * cy + wr[o][5] * cz;
            s = lrelu(gg[o] * s + bbv[o]);
            acc[o] = fmaxf(acc[o], s);
        }
    }
#pragma unroll
    for (int o = 0; o < 8; ++o)
        cat[(size_t)p * CAT_C + og + o] = (_Float16)acc[o];
}

// ---------------------------------------------------------------------------
// Graph max-pool: pooled[p, 0:ci] = max_k cat[b, idx[p,k], off:off+ci].
// One thread per (point, 8-channel chunk); 16B vector gathers.
// ---------------------------------------------------------------------------
__global__ void pool_max(const _Float16* __restrict__ cat, const long long* __restrict__ idx,
                         _Float16* __restrict__ pooled, int ci, int chanOffPrev)
{
    int chunks = ci >> 3;
    int t = blockIdx.x * blockDim.x + threadIdx.x;
    if (t >= BB * NN * chunks) return;
    int p  = t / chunks;
    int cc = (t % chunks) << 3;
    int b  = p / NN;
    const long long* ip = idx + (size_t)p * KK;
    float m[8];
#pragma unroll
    for (int e = 0; e < 8; ++e) m[e] = -1e30f;
    for (int k = 0; k < KK; ++k) {
        int j = (int)ip[k];
        union { uint4 u; _Float16 h[8]; } v;
        v.u = *(const uint4*)(cat + ((size_t)b * NN + j) * CAT_C + chanOffPrev + cc);
#pragma unroll
        for (int e = 0; e < 8; ++e) m[e] = fmaxf(m[e], (float)v.h[e]);
    }
    _Float16* d = pooled + (size_t)p * ci + cc;
#pragma unroll
    for (int e = 0; e < 8; ++e) d[e] = (_Float16)m[e];
}

// ---------------------------------------------------------------------------
// WMMA GEMM + bias + LeakyReLU. One wave per 16x16 output tile.
// A: (B*N, lda) f16 row-major.  Bp: pre-packed fragments.  Writes f16 into
// out[:, chanOff:chanOff+co] with row stride ldo.
// A fragment (16-bit A 16x32): lane L = row m=(L&15); halves 0..7 hold
// K = kt*32 + (L>>4)*8 + j, halves 8..15 hold K = kt*32 + 16 + (L>>4)*8 + j.
// ---------------------------------------------------------------------------
__global__ __launch_bounds__(32) void gemm_bn_lrelu(
    const _Float16* __restrict__ A, int lda,
    const _Float16* __restrict__ Bp,
    const float* __restrict__ bias,
    _Float16* __restrict__ out, int ldo, int chanOff,
    int ci, int co)
{
    const int rowTile = blockIdx.x;
    const int ct      = blockIdx.y;
    const int lane    = threadIdx.x;
    const int h16     = lane >> 4;
    const int l16     = lane & 15;
    const int numKT   = ci >> 5;
    const int numCT   = co >> 4;
    const int rowBase = rowTile << 4;

    const _Float16* arow = A + (size_t)(rowBase + l16) * lda;
    const _Float16* bptr = Bp + ((size_t)ct * 32 + lane) * 16;
    const size_t bstride = (size_t)numCT * 512;   // halves per kt step

    v8f acc = {};
    for (int kt = 0; kt < numKT; ++kt) {
        union { uint4 u[2]; v16h v; } a;
        a.u[0] = *(const uint4*)(arow + (kt << 5) + h16 * 8);
        a.u[1] = *(const uint4*)(arow + (kt << 5) + 16 + h16 * 8);
        v16h bfrag = *(const v16h*)bptr;
        __builtin_prefetch((const void*)(bptr + bstride), 0, 0);
        acc = __builtin_amdgcn_wmma_f32_16x16x32_f16(
                  false, a.v, false, bfrag, (short)0, acc, false, false);
        bptr += bstride;
    }
    const float be = bias[(ct << 4) + l16];
#pragma unroll
    for (int j = 0; j < 8; ++j) {
        float v = lrelu(acc[j] + be);
        int m = (h16 << 3) + j;
        out[(size_t)(rowBase + m) * ldo + chanOff + (ct << 4) + l16] = (_Float16)v;
    }
}

// ---------------------------------------------------------------------------
// Final GEMM (512->512) + bias + LReLU + column-max over the 16-row tile.
// Writes per-row-tile partial maxes; reduce_max finishes max over N.
// ---------------------------------------------------------------------------
__global__ __launch_bounds__(32) void final_gemm_colmax(
    const _Float16* __restrict__ A,      // cat, lda = 512
    const _Float16* __restrict__ Bp,     // packed w_fin (512x512)
    const float* __restrict__ bias,
    float* __restrict__ partial)         // (B*N/16, 512)
{
    const int rowTile = blockIdx.x;      // 0 .. B*N/16 - 1
    const int ct      = blockIdx.y;      // 0 .. 31
    const int lane    = threadIdx.x;
    const int h16     = lane >> 4;
    const int l16     = lane & 15;
    const int rowBase = rowTile << 4;
    const int numKT   = CAT_C >> 5;      // 16
    const int numCT   = 32;

    const _Float16* arow = A + (size_t)(rowBase + l16) * CAT_C;
    const _Float16* bptr = Bp + ((size_t)ct * 32 + lane) * 16;
    const size_t bstride = (size_t)numCT * 512;

    v8f acc = {};
#pragma unroll 4
    for (int kt = 0; kt < numKT; ++kt) {
        union { uint4 u[2]; v16h v; } a;
        a.u[0] = *(const uint4*)(arow + (kt << 5) + h16 * 8);
        a.u[1] = *(const uint4*)(arow + (kt << 5) + 16 + h16 * 8);
        v16h bfrag = *(const v16h*)bptr;
        __builtin_prefetch((const void*)(bptr + bstride), 0, 0);
        acc = __builtin_amdgcn_wmma_f32_16x16x32_f16(
                  false, a.v, false, bfrag, (short)0, acc, false, false);
        bptr += bstride;
    }
    const float be = bias[(ct << 4) + l16];
    float cm = -1e30f;
#pragma unroll
    for (int j = 0; j < 8; ++j) cm = fmaxf(cm, lrelu(acc[j] + be));
    cm = fmaxf(cm, __shfl_xor(cm, 16, 32));   // combine rows 0-7 with 8-15 (wave32)
    if (lane < 16)
        partial[(size_t)rowTile * 512 + (ct << 4) + l16] = cm;
}

__global__ void reduce_max(const float* __restrict__ partial, float* __restrict__ out)
{
    int t = blockIdx.x * blockDim.x + threadIdx.x;
    if (t >= BB * 512) return;
    int b = t >> 9, o = t & 511;
    const int RT = NN / 16;
    float m = -1e30f;
    for (int rt = 0; rt < RT; ++rt)
        m = fmaxf(m, partial[((size_t)b * RT + rt) * 512 + o]);
    out[t] = m;
}

// ---------------------------------------------------------------------------
extern "C" void kernel_launch(void* const* d_in, const int* in_sizes, int n_in,
                              void* d_out, int out_size, void* d_ws, size_t ws_size,
                              hipStream_t stream)
{
    (void)in_sizes; (void)n_in; (void)out_size; (void)ws_size;
    const float*     x      = (const float*)d_in[0];
    const long long* idx    = (const long long*)d_in[1];   // jnp.int64
    const float*     w_edge = (const float*)d_in[2];
    const float*     g_edge = (const float*)d_in[3];
    const float*     be_edge= (const float*)d_in[4];
    const float*     w0 = (const float*)d_in[5],  *g0 = (const float*)d_in[6],  *be0 = (const float*)d_in[7];
    const float*     w1 = (const float*)d_in[8],  *g1 = (const float*)d_in[9],  *be1 = (const float*)d_in[10];
    const float*     w2 = (const float*)d_in[11], *g2 = (const float*)d_in[12], *be2 = (const float*)d_in[13];
    const float*     w_fin = (const float*)d_in[14];
    const float*     b_fin = (const float*)d_in[15];
    float* out = (float*)d_out;

    // Workspace carving (all offsets 256B-aligned)
    char* ws = (char*)d_ws;
    _Float16* cat    = (_Float16*)ws;  ws += (size_t)BB * NN * CAT_C * 2;   // 33.5 MB
    _Float16* pooled = (_Float16*)ws;  ws += (size_t)BB * NN * 128 * 2;     //  8.4 MB
    _Float16* pw0    = (_Float16*)ws;  ws += (size_t)64  * 64  * 2;
    _Float16* pw1    = (_Float16*)ws;  ws += (size_t)128 * 64  * 2;
    _Float16* pw2    = (_Float16*)ws;  ws += (size_t)256 * 128 * 2;
    _Float16* pwf    = (_Float16*)ws;  ws += (size_t)512 * 512 * 2;
    float*    part   = (float*)ws;     ws += (size_t)BB * (NN / 16) * 512 * 4; // 4.2 MB

    // Pack BN-folded weights into WMMA B fragments
    pack_weight<<<(64*64   + 255) / 256, 256, 0, stream>>>(w0,    g0,      pw0, 64,  64);
    pack_weight<<<(128*64  + 255) / 256, 256, 0, stream>>>(w1,    g1,      pw1, 64,  128);
    pack_weight<<<(256*128 + 255) / 256, 256, 0, stream>>>(w2,    g2,      pw2, 128, 256);
    pack_weight<<<(512*512 + 255) / 256, 256, 0, stream>>>(w_fin, nullptr, pwf, 512, 512);

    // Stage 1: EdgeConv -> cat[:, 0:64]
    edge_conv<<<(BB*NN*8 + 255) / 256, 256, 0, stream>>>(x, idx, w_edge, g_edge, be_edge, cat);

    const int rowTiles = BB * NN / 16;   // 2048

    // Layer 1: pool h0 (off 0, ci 64) -> GEMM 64->64 -> cat[:, 64:128]
    pool_max<<<(BB*NN*8 + 255) / 256, 256, 0, stream>>>(cat, idx, pooled, 64, 0);
    gemm_bn_lrelu<<<dim3(rowTiles, 4), 32, 0, stream>>>(pooled, 64, pw0, be0, cat, CAT_C, 64, 64, 64);

    // Layer 2: pool h1 (off 64, ci 64) -> GEMM 64->128 -> cat[:, 128:256]
    pool_max<<<(BB*NN*8 + 255) / 256, 256, 0, stream>>>(cat, idx, pooled, 64, 64);
    gemm_bn_lrelu<<<dim3(rowTiles, 8), 32, 0, stream>>>(pooled, 64, pw1, be1, cat, CAT_C, 128, 64, 128);

    // Layer 3: pool h2 (off 128, ci 128) -> GEMM 128->256 -> cat[:, 256:512]
    pool_max<<<(BB*NN*16 + 255) / 256, 256, 0, stream>>>(cat, idx, pooled, 128, 128);
    gemm_bn_lrelu<<<dim3(rowTiles, 16), 32, 0, stream>>>(pooled, 128, pw2, be2, cat, CAT_C, 256, 128, 256);

    // Final: 512->512 GEMM + bias + LReLU + max over N (two stages)
    final_gemm_colmax<<<dim3(rowTiles, 32), 32, 0, stream>>>(cat, pwf, b_fin, part);
    reduce_max<<<(BB*512 + 255) / 256, 256, 0, stream>>>(part, out);
}
